// Refiner_32770600468873
// MI455X (gfx1250) — compile-verified, matrix operand order checked by reading, not run on hardware
//
#include <hip/hip_runtime.h>
#include <hip/hip_bf16.h>
#include <math.h>

typedef _Float16 v16h __attribute__((ext_vector_type(16)));
typedef float    v8f  __attribute__((ext_vector_type(8)));

#define BN_EPS 1e-5f

// ---------------------------------------------------------------------------
// Fragment-layout helper: element j (0..15) of lane L holds
//   row/col = L % 16
//   k_local = (L>=16 ? 8 : 0) + (j < 8 ? j : j + 8)        (K-block of 32)
// matching the CDNA5 16-bit A/B fragment VGPR layout (ISA 7.12.2).
// ---------------------------------------------------------------------------
__device__ __forceinline__ int frag_klocal(int lane, int j) {
    return ((lane >= 16) ? 8 : 0) + ((j < 8) ? j : j + 8);
}

// ---------------------------------------------------------------------------
// Edge-clamped bilinear sample on an n x n f32 image (half-pixel convention).
// ---------------------------------------------------------------------------
__device__ __forceinline__ float bilerp(const float* img, int n, float fy, float fx) {
    float yf = floorf(fy), xf = floorf(fx);
    int y0 = (int)yf, x0 = (int)xf;
    float wy = fy - yf, wx = fx - xf;
    int y0c = min(max(y0, 0), n - 1);
    int y1c = min(max(y0 + 1, 0), n - 1);
    int x0c = min(max(x0, 0), n - 1);
    int x1c = min(max(x0 + 1, 0), n - 1);
    float v00 = img[y0c * n + x0c], v01 = img[y0c * n + x1c];
    float v10 = img[y1c * n + x0c], v11 = img[y1c * n + x1c];
    float top = v00 + wx * (v01 - v00);
    float bot = v10 + wx * (v11 - v10);
    return top + wy * (bot - top);
}

// ---------------------------------------------------------------------------
// GEMM #1 operand packing, fragment-major.
// A: feat tokens, per (stage, mtile, kstep):  [4][13][24][32 lanes][16 halves]
//    A[q][c] = fv[stage][q+1][c], rows q>=196 zero-padded.
// ---------------------------------------------------------------------------
__global__ void pack_Afrag(const float* __restrict__ fv, _Float16* __restrict__ Ap) {
    int idx = blockIdx.x * blockDim.x + threadIdx.x;
    if (idx >= 4 * 13 * 24 * 512) return;
    int j    = idx & 15;
    int lane = (idx >> 4) & 31;
    int ks   = (idx >> 9) % 24;
    int rest = (idx >> 9) / 24;
    int tile = rest % 13;
    int stage = rest / 13;
    int q = tile * 16 + (lane & 15);
    int k = ks * 32 + frag_klocal(lane, j);
    float v = (q < 196) ? fv[stage * 197 * 768 + (q + 1) * 768 + k] : 0.0f;
    Ap[idx] = (_Float16)v;
}

// B: conv tap weights, per kstep: [24][32][16];  B[c][t] = conv_w[0][1+c][t],
// taps t>=9 zero-padded (N padded 9 -> 16).
__global__ void pack_Bfrag(const float* __restrict__ conv_w, _Float16* __restrict__ Bp) {
    int idx = blockIdx.x * blockDim.x + threadIdx.x;
    if (idx >= 24 * 512) return;
    int j    = idx & 15;
    int lane = (idx >> 4) & 31;
    int ks   = idx >> 9;
    int t = lane & 15;                       // N (tap)
    int c = ks * 32 + frag_klocal(lane, j);  // K (channel)
    float v = (t < 9) ? conv_w[(1 + c) * 9 + t] : 0.0f;
    Bp[idx] = (_Float16)v;
}

// ---------------------------------------------------------------------------
// WMMA GEMM #1: G[s][t][q] = sum_c feat[s][q][c] * w[c][t]
// 52 waves (4 stages x 13 M-tiles), K = 768 (24 steps).
// Inner loop: 2 fragment loads (2x b128 each) + 1 wmma.
// ---------------------------------------------------------------------------
__global__ void gemm_G(const _Float16* __restrict__ Ap,
                       const _Float16* __restrict__ Bp,
                       float* __restrict__ G) {
    int wave  = blockIdx.x;              // 0..51
    int stage = wave / 13;
    int tile  = wave - stage * 13;
    int lane  = threadIdx.x & 31;
    const v16h* Af = (const v16h*)(Ap + (size_t)wave * 24 * 512) + lane;
    const v16h* Bf = (const v16h*)Bp + lane;

    v8f c = {};
#pragma unroll 4
    for (int ks = 0; ks < 24; ++ks) {
        v16h a = Af[ks * 32];
        v16h b = Bf[ks * 32];
        c = __builtin_amdgcn_wmma_f32_16x16x32_f16(false, a, false, b,
                                                   (short)0, c, false, false);
    }
    int col = lane & 15;
#pragma unroll
    for (int r = 0; r < 8; ++r) {
        int m = ((lane >= 16) ? 8 : 0) + r;  // D: M = vgpr + (lane>=16 ? 8:0)
        int q = tile * 16 + m;
        if (q < 196 && col < 9)
            G[stage * 9 * 196 + col * 196 + q] = c[r];
    }
}

// ---------------------------------------------------------------------------
// Fused stage: conv3x3 over [h_in(upsampled Rp->R) ; feat(upsampled)] + bias,
// ReLU, BatchNorm(eval). Feature term = bilinear sample of precomputed G_t on
// the 14x14 base grid (exact algebraic refactoring of the reference).
// ---------------------------------------------------------------------------
__global__ void conv_stage(int R, int Rp,
                           const float* __restrict__ hprev,  // Rp x Rp
                           const float* __restrict__ G,      // [9][196]
                           const float* __restrict__ conv_w, // [769*9]
                           const float* __restrict__ conv_b,
                           const float* __restrict__ bn_g,
                           const float* __restrict__ bn_b,
                           const float* __restrict__ bn_m,
                           const float* __restrict__ bn_v,
                           float* __restrict__ hout) {       // R x R
    int idx = blockIdx.x * blockDim.x + threadIdx.x;
    if (idx >= R * R) return;
    int y = idx / R, x = idx - y * R;
    float acc = conv_b[0];
    float sh = (float)Rp / (float)R;   // h-channel src scale
    float sf = 14.0f / (float)R;       // feature (G) src scale
#pragma unroll
    for (int ty = 0; ty < 3; ++ty) {
#pragma unroll
        for (int tx = 0; tx < 3; ++tx) {
            int yy = y + ty - 1, xx = x + tx - 1;
            if (yy < 0 || yy >= R || xx < 0 || xx >= R) continue; // SAME zero pad
            int t = ty * 3 + tx;
            float hy = (yy + 0.5f) * sh - 0.5f;
            float hx = (xx + 0.5f) * sh - 0.5f;
            acc += conv_w[t] * bilerp(hprev, Rp, hy, hx);        // h channel
            float fy = (yy + 0.5f) * sf - 0.5f;
            float fx = (xx + 0.5f) * sf - 0.5f;
            acc += bilerp(G + t * 196, 14, fy, fx);              // 768 feat chans
        }
    }
    acc = fmaxf(acc, 0.0f);
    float sc = bn_g[0] * rsqrtf(bn_v[0] + BN_EPS);
    hout[idx] = acc * sc + (bn_b[0] - bn_m[0] * sc);
}

// ---------------------------------------------------------------------------
// GEMM #2 operand packing.
// A: bilinear x2 upsample of h4 (112->224), written directly in A-fragment
//    layout [14 mtiles][7 ksteps][32][16] (bijection onto 224x224 image).
// ---------------------------------------------------------------------------
__global__ void pack_Hfrag(const float* __restrict__ h112, _Float16* __restrict__ Hf) {
    int idx = blockIdx.x * blockDim.x + threadIdx.x;
    if (idx >= 14 * 7 * 512) return;
    int j    = idx & 15;
    int lane = (idx >> 4) & 31;
    int ks   = (idx >> 9) % 7;
    int tile = (idx >> 9) / 7;
    int row = tile * 16 + (lane & 15);       // M (output row)
    int k   = ks * 32 + frag_klocal(lane, j);// K (w coordinate)
    float fy = (row + 0.5f) * 0.5f - 0.5f;
    float fx = (k + 0.5f) * 0.5f - 0.5f;
    Hf[idx] = (_Float16)bilerp(h112, 112, fy, fx);
}

// B: W_w accessed transposed — B[k][n] = W_w[n][k]; layout [14 ntiles][7][32][16].
__global__ void pack_Wfrag(const float* __restrict__ Ww, _Float16* __restrict__ Wf) {
    int idx = blockIdx.x * blockDim.x + threadIdx.x;
    if (idx >= 14 * 7 * 512) return;
    int j    = idx & 15;
    int lane = (idx >> 4) & 31;
    int ks   = (idx >> 9) % 7;
    int tile = (idx >> 9) / 7;
    int n = tile * 16 + (lane & 15);          // N (output feature o)
    int k = ks * 32 + frag_klocal(lane, j);   // K (w coordinate)
    Wf[idx] = (_Float16)Ww[n * 224 + k];
}

// ---------------------------------------------------------------------------
// WMMA GEMM #2 + epilogue: out[r][o] = sigmoid( sum_w H[r][w]*Ww[o][w] + Wb[o] )
// M = N = 224 (14x14 tiles), K = 224 (7 steps). 196 waves, 49 blocks x 128.
// ---------------------------------------------------------------------------
__global__ void final_gemm(const _Float16* __restrict__ Hf,
                           const _Float16* __restrict__ Wf,
                           const float* __restrict__ Wb,
                           float* __restrict__ out) {
    int wave   = blockIdx.x * (blockDim.x >> 5) + (threadIdx.x >> 5);
    int tile_m = wave / 14, tile_n = wave - tile_m * 14;
    int lane   = threadIdx.x & 31;
    const v16h* Af = (const v16h*)(Hf + (size_t)tile_m * 7 * 512) + lane;
    const v16h* Bf = (const v16h*)(Wf + (size_t)tile_n * 7 * 512) + lane;

    v8f c = {};
#pragma unroll
    for (int ks = 0; ks < 7; ++ks) {
        v16h a = Af[ks * 32];
        v16h b = Bf[ks * 32];
        c = __builtin_amdgcn_wmma_f32_16x16x32_f16(false, a, false, b,
                                                   (short)0, c, false, false);
    }
    int col = tile_n * 16 + (lane & 15);
    float bias = Wb[col];
#pragma unroll
    for (int r = 0; r < 8; ++r) {
        int m = tile_m * 16 + ((lane >= 16) ? 8 : 0) + r;
        float v = c[r] + bias;
        out[m * 224 + col] = 1.0f / (1.0f + __expf(-v));
    }
}

// ---------------------------------------------------------------------------
extern "C" void kernel_launch(void* const* d_in, const int* in_sizes, int n_in,
                              void* d_out, int out_size, void* d_ws, size_t ws_size,
                              hipStream_t stream) {
    const float* x      = (const float*)d_in[0];   // [14,14]
    const float* fv     = (const float*)d_in[1];   // [4,197,768]
    const float* conv_w = (const float*)d_in[2];   // [1,769,3,3]
    const float* conv_b = (const float*)d_in[3];
    const float* bn_g   = (const float*)d_in[4];
    const float* bn_b   = (const float*)d_in[5];
    const float* bn_m   = (const float*)d_in[6];
    const float* bn_v   = (const float*)d_in[7];
    const float* W_w    = (const float*)d_in[8];   // [224,224]
    const float* W_b    = (const float*)d_in[9];   // [224]
    float* out          = (float*)d_out;           // [224*224]

    char* ws = (char*)d_ws;
    size_t off = 0;
    auto carve = [&](size_t bytes) {
        char* p = ws + off;
        off = (off + bytes + 255) & ~(size_t)255;
        return p;
    };
    const int NA  = 4 * 13 * 24 * 512;   // A fragments for gemm_G
    const int NB  = 24 * 512;            // B fragments for gemm_G
    const int NHW = 14 * 7 * 512;        // fragments for final gemm (each side)
    _Float16* Ap = (_Float16*)carve((size_t)NA * sizeof(_Float16));
    _Float16* Bp = (_Float16*)carve((size_t)NB * sizeof(_Float16));
    _Float16* Hf = (_Float16*)carve((size_t)NHW * sizeof(_Float16));
    _Float16* Wf = (_Float16*)carve((size_t)NHW * sizeof(_Float16));
    float*    G  = (float*)carve(4 * 9 * 196 * sizeof(float));
    float*    h1 = (float*)carve(14 * 14 * sizeof(float));
    float*    h2 = (float*)carve(28 * 28 * sizeof(float));
    float*    h3 = (float*)carve(56 * 56 * sizeof(float));
    float*    h4 = (float*)carve(112 * 112 * sizeof(float));
    (void)in_sizes; (void)n_in; (void)out_size; (void)ws_size;

    pack_Afrag<<<(NA + 255) / 256, 256, 0, stream>>>(fv, Ap);
    pack_Bfrag<<<(NB + 255) / 256, 256, 0, stream>>>(conv_w, Bp);
    pack_Wfrag<<<(NHW + 255) / 256, 256, 0, stream>>>(W_w, Wf);

    gemm_G<<<52, 32, 0, stream>>>(Ap, Bp, G);  // G[s][t][q] for all 4 stages

    const int G9 = 9 * 196;
    // Stage order uses fv[3], fv[2], fv[1], fv[0].
    conv_stage<<<(14 * 14 + 127) / 128, 128, 0, stream>>>(
        14, 14, x, G + 3 * G9, conv_w, conv_b, bn_g, bn_b, bn_m, bn_v, h1);
    conv_stage<<<(28 * 28 + 127) / 128, 128, 0, stream>>>(
        28, 14, h1, G + 2 * G9, conv_w, conv_b, bn_g, bn_b, bn_m, bn_v, h2);
    conv_stage<<<(56 * 56 + 127) / 128, 128, 0, stream>>>(
        56, 28, h2, G + 1 * G9, conv_w, conv_b, bn_g, bn_b, bn_m, bn_v, h3);
    conv_stage<<<(112 * 112 + 127) / 128, 128, 0, stream>>>(
        112, 56, h3, G + 0 * G9, conv_w, conv_b, bn_g, bn_b, bn_m, bn_v, h4);

    pack_Hfrag<<<(NHW + 255) / 256, 256, 0, stream>>>(h4, Hf);

    final_gemm<<<49, 128, 0, stream>>>(Hf, Wf, W_b, out);
}
